// rnnLSTM_2568390443243
// MI455X (gfx1250) — matrix operand order repaired and checked by reading, not compiled, not run on hardware
//
#include <hip/hip_runtime.h>
#include <hip/hip_bf16.h>
#include <math.h>

typedef __bf16 bf16;
typedef __attribute__((ext_vector_type(16))) __bf16 v16bf;
typedef __attribute__((ext_vector_type(8)))  __bf16 v8bf;
typedef __attribute__((ext_vector_type(8)))  float  v8f;

#define T_STEPS 512
#define IN_SZ   10
#define HID     100
#define G4      400       // 4*HID
#define KPAD    128       // HID + IN_SZ padded to multiple of 32
#define BTILE   16
#define GT_W    17        // padded row width (odd -> conflict-free strided reads)
#define NCLS    14
#define NTHREADS 256
#define H24     84        // h index where gate tile 24 starts (o-gate cols 384..399)

__device__ __forceinline__ float dev_tanh(float v) {
#if __has_builtin(__builtin_amdgcn_tanhf)
    return __builtin_amdgcn_tanhf(v);           // v_tanh_f32 (CDNA5 TRANS op)
#else
    float e = __builtin_amdgcn_exp2f(v * -2.8853900817779268f);   // exp(-2v)
    return (1.0f - e) * __builtin_amdgcn_rcpf(1.0f + e);
#endif
}
__device__ __forceinline__ float dev_sigmoid(float v) {
    return __builtin_fmaf(dev_tanh(0.5f * v), 0.5f, 0.5f);
}

// One workgroup owns a 16-row batch tile for the whole sequence.
// gates = [h | x_t] (16x128 bf16, LDS) @ Wc^T (400x128 bf16, registers) via WMMA.
// __launch_bounds__(256, 1): occupancy is grid-limited (128 WGs ~ 1/WGP), so let
// the RA keep all 10 A-fragment loads in flight instead of recycling registers.
__global__ __launch_bounds__(NTHREADS, 1) void lstm_wmma_kernel(
    const float* __restrict__ x,
    const float* __restrict__ w_ih,
    const float* __restrict__ w_hh,
    const float* __restrict__ b_ih,
    const float* __restrict__ b_hh,
    const float* __restrict__ fc_w,
    const float* __restrict__ fc_b,
    float* __restrict__ out)
{
    __shared__ __align__(32) bf16  Wc[G4][KPAD];        // 102400 B: [w_hh | w_ih | 0]
    __shared__ float bias[G4];                          //   1600 B: b_ih + b_hh
    __shared__ __align__(32) bf16  Abuf[BTILE][KPAD];   //   4096 B: [h | x_t | 0]
    __shared__ float gatesT[G4][GT_W];                  //  27200 B: column-major gates (padded)
    __shared__ float p24[4][16][GT_W];                  //   4352 B: tile-24 K-partials
    __shared__ float cbuf[BTILE][HID];                  //   6400 B
    __shared__ float hbuf[BTILE][HID];                  //   6400 B   (total ~149 KB)

    const int tid  = threadIdx.x;
    const int lane = tid & 31;
    const int wave = tid >> 5;
    const int swave = __builtin_amdgcn_readfirstlane(wave);  // scalar wave id
    const int hsel = lane >> 4;    // which 16-lane half of the wave
    const int mr   = lane & 15;
    const int b0   = blockIdx.x * BTILE;

    // x staging assignment (160 threads cover the 16x10 x_t tile)
    const bool xactive = tid < BTILE * IN_SZ;
    const int  xr = tid / IN_SZ;
    const int  xc = tid - xr * IN_SZ;

    // ---- one-time setup ----
    for (int i = tid; i < G4 * KPAD; i += NTHREADS) {
        int g = i >> 7, k = i & (KPAD - 1);
        float v = 0.0f;
        if (k < HID)              v = w_hh[g * HID + k];
        else if (k < HID + IN_SZ) v = w_ih[g * IN_SZ + (k - HID)];
        Wc[g][k] = (bf16)v;
    }
    for (int g = tid; g < G4; g += NTHREADS) bias[g] = b_ih[g] + b_hh[g];
    for (int i = tid; i < BTILE * KPAD; i += NTHREADS) {
        int k = i & (KPAD - 1);
        if (k < HID || k >= HID + IN_SZ) ((bf16*)Abuf)[i] = (bf16)0.0f;  // skip x slot
    }
    for (int i = tid; i < BTILE * HID; i += NTHREADS) ((float*)cbuf)[i] = 0.0f;

    float xreg = 0.0f;
    if (xactive) {
        const size_t rowbase = (size_t)(b0 + xr) * T_STEPS;
        Abuf[xr][HID + xc] = (bf16)x[(rowbase + 0) * IN_SZ + xc];   // x_0
        xreg = x[(rowbase + 1) * IN_SZ + xc];                       // prefetch x_1
    }
    __syncthreads();

    // ---- hoist this wave's B fragments (weights) into registers: loop-invariant ----
    // B 32x16 bf16 layout: N = lane&15, K half from lane half, 32 contiguous bytes per lane.
    const bf16* wr0 = &Wc[(swave      ) * 16 + mr][0];
    const bf16* wr1 = &Wc[(swave +  8) * 16 + mr][0];
    const bf16* wr2 = &Wc[(swave + 16) * 16 + mr][0];
    v16bf breg0[4], breg1[4], breg2[4];
    #pragma unroll
    for (int kt = 0; kt < 4; ++kt) {
        const int off = kt * 32 + hsel * 16;
        breg0[kt] = *(const v16bf*)(wr0 + off);
        breg1[kt] = *(const v16bf*)(wr1 + off);
        breg2[kt] = *(const v16bf*)(wr2 + off);
    }
    // waves 4..7 each own one K-chunk of tile 24 (one extra B fragment)
    const int kt24 = (swave >= 4) ? (swave - 4) : 0;         // clamped: valid for all waves
    v16bf breg3 = {};
    if (swave >= 4)
        breg3 = *(const v16bf*)(&Wc[24 * 16 + mr][0] + kt24 * 32 + hsel * 16);

    // loop-invariant addresses
    const int   k24base = kt24 * 32 + hsel * 16;             // tile-24 A-fragment column base
    const bf16* arow    = &Abuf[mr][0];
    float* const d0 = &gatesT[(swave      ) * 16 + mr][hsel * 8];
    float* const d1 = &gatesT[(swave +  8) * 16 + mr][hsel * 8];
    float* const d2 = &gatesT[(swave + 16) * 16 + mr][hsel * 8];
    float* const dp = &p24[kt24][mr][hsel * 8];

    for (int t = 0; t < T_STEPS; ++t) {
        // ---- WMMA phase ----
        // A fragments (16x32 bf16 layout): lane half 0 -> K {kb..kb+15}, half 1 -> +16
        v16bf afrag[4];
        #pragma unroll
        for (int kt = 0; kt < 4; ++kt) {
            const int kb = kt * 32 + hsel * 16;
            v8bf lo = *(const v8bf*)(arow + kb);
            v8bf hi = *(const v8bf*)(arow + kb + 8);
            #pragma unroll
            for (int e = 0; e < 8; ++e) { afrag[kt][e] = lo[e]; afrag[kt][e + 8] = hi[e]; }
        }
        // tile-24 A fragment re-loaded via uniform per-wave address (no register-array select)
        v16bf af24;
        {
            v8bf lo = *(const v8bf*)(arow + k24base);
            v8bf hi = *(const v8bf*)(arow + k24base + 8);
            #pragma unroll
            for (int e = 0; e < 8; ++e) { af24[e] = lo[e]; af24[e + 8] = hi[e]; }
        }

        // three independent accumulator chains per wave (tiles w, w+8, w+16), B from registers
        v8f a0 = {}, a1 = {}, a2 = {};
        #pragma unroll
        for (int kt = 0; kt < 4; ++kt) {
            a0 = __builtin_amdgcn_wmma_f32_16x16x32_bf16(false, afrag[kt], false, breg0[kt], (short)0, a0, false, false);
            a1 = __builtin_amdgcn_wmma_f32_16x16x32_bf16(false, afrag[kt], false, breg1[kt], (short)0, a1, false, false);
            a2 = __builtin_amdgcn_wmma_f32_16x16x32_bf16(false, afrag[kt], false, breg2[kt], (short)0, a2, false, false);
        }
        // tile 24: one K-chunk per wave 4..7 -> partial buffers (summed in activation)
        if (swave >= 4) {
            v8f a3 = {};
            a3 = __builtin_amdgcn_wmma_f32_16x16x32_bf16(false, af24, false, breg3, (short)0, a3, false, false);
            #pragma unroll
            for (int j = 0; j < 8; ++j) dp[j] = a3[j];
        }
        {   // C/D layout: lane holds N=mr, M = hsel*8 + j
            #pragma unroll
            for (int j = 0; j < 8; ++j) { d0[j] = a0[j]; d1[j] = a1[j]; d2[j] = a2[j]; }
        }
        __syncthreads();

        // ---- activation / state-update phase ----
        // stage x_{t+1} (already in xreg) and prefetch x_{t+2}
        if (xactive) {
            Abuf[xr][HID + xc] = (bf16)xreg;
            const int tn = (t + 2 < T_STEPS) ? t + 2 : T_STEPS - 1;
            xreg = x[((size_t)(b0 + xr) * T_STEPS + tn) * IN_SZ + xc];
        }
        for (int idx = tid; idx < BTILE * HID; idx += NTHREADS) {
            int b = idx / HID, h = idx - b * HID;
            float gi = dev_sigmoid(gatesT[h][b]           + bias[h]);
            float gf = dev_sigmoid(gatesT[HID + h][b]     + bias[HID + h]);
            float gg = dev_tanh   (gatesT[2 * HID + h][b] + bias[2 * HID + h]);
            float go_in;
            if (h >= H24) {           // o-gate cols 384..399 come from the 4 K-partials
                const int r = h - H24;
                go_in = (p24[0][r][b] + p24[1][r][b]) + (p24[2][r][b] + p24[3][r][b]);
            } else {
                go_in = gatesT[3 * HID + h][b];
            }
            float go = dev_sigmoid(go_in + bias[3 * HID + h]);
            float c = gf * cbuf[b][h] + gi * gg;
            cbuf[b][h] = c;
            float hv = go * dev_tanh(c);
            hbuf[b][h] = hv;
            Abuf[b][h] = (bf16)hv;   // next step's A matrix (h part)
        }
        __syncthreads();
    }

    // ---- final FC: out[b] = hT @ fc_w^T + fc_b  (tiny, VALU) ----
    for (int idx = tid; idx < BTILE * NCLS; idx += NTHREADS) {
        int r = idx / NCLS, cls = idx - r * NCLS;
        float acc = fc_b[cls];
        #pragma unroll 4
        for (int h = 0; h < HID; ++h) acc += hbuf[r][h] * fc_w[cls * HID + h];
        out[(size_t)(b0 + r) * NCLS + cls] = acc;
    }
}

extern "C" void kernel_launch(void* const* d_in, const int* in_sizes, int n_in,
                              void* d_out, int out_size, void* d_ws, size_t ws_size,
                              hipStream_t stream) {
    const float* x    = (const float*)d_in[0];
    const float* w_ih = (const float*)d_in[1];
    const float* w_hh = (const float*)d_in[2];
    const float* b_ih = (const float*)d_in[3];
    const float* b_hh = (const float*)d_in[4];
    const float* fc_w = (const float*)d_in[5];
    const float* fc_b = (const float*)d_in[6];
    float* out = (float*)d_out;

    int Btotal = in_sizes[0] / (T_STEPS * IN_SZ);   // 2048
    int grid = Btotal / BTILE;                      // 128 workgroups
    lstm_wmma_kernel<<<grid, NTHREADS, 0, stream>>>(
        x, w_ih, w_hh, b_ih, b_hh, fc_w, fc_b, out);
}
